// SPEMDescriptor_45500883534093
// MI455X (gfx1250) — matrix operand order repaired and checked by reading, not compiled, not run on hardware
//
#include <hip/hip_runtime.h>
#include <hip/hip_bf16.h>
#include <math.h>

typedef __attribute__((ext_vector_type(16))) _Float16 v16h;
typedef __attribute__((ext_vector_type(8)))  float    v8f;
typedef __attribute__((ext_vector_type(4)))  _Float16 v4h;

#define IMG   512
#define NB    16
#define NCH   12
#define TILE  16
#define HALO  6
#define RW    28          // tile row width  = TILE + 2*HALO
#define RH    29          // tile rows: 28 real + 1 zero row (dy can reach 13)
#define KPAD  224         // K = dy*16 + dx, dy in 0..13, dx in 0..15 (zero padded)
#define NQ    7           // 224 / 32 K-chunks

// ---------------------------------------------------------------------------
// K1: build composite 13x13 filter bank (full conv of the two 7x7 banks),
//     pack WMMA A-matrix f16 [16][224] with k = dy*16+dx, zero stats buffer.
// ---------------------------------------------------------------------------
__global__ __launch_bounds__(256)
void init_kernel(_Float16* __restrict__ Apack, float* __restrict__ stats)
{
    __shared__ float W12[12][7][7];
    __shared__ float G7[7][7];
    const int tid = threadIdx.x;

    for (int i = tid; i < NB * NCH * 2; i += 256) stats[i] = 0.f;

    const double PI = 3.14159265358979323846;

    // ---- first/second order Gaussian-derivative bank, summed over sigmas ----
    for (int e = tid; e < 12 * 49; e += 256) {
        const int c = e / 49, pos = e % 49;
        const int ai = pos / 7, bi = pos % 7;
        const double x = ai - 3, y = bi - 3;           // meshgrid 'ij'
        const double th = (double)(c % 6) * PI / 5.0;  // linspace(0, pi, 6)
        const double sig[3] = {0.6, 0.8, 1.0};
        double acc = 0.0;
        for (int si = 0; si < 3; ++si) {
            const double s = sig[si];
            const int size = ((int)(6.0 * s)) | 1;     // 3, 5, 7
            const int h = size / 2;
            if (x < -h || x > h || y < -h || y > h) continue;   // zero-embed
            const double r2 = x * x + y * y;
            if (c < 6) {
                const double Gx = -x * exp(-r2 / (2.0 * s * s)) / (2.0 * PI * s * s * s * s);
                // faithful to reference: Gy exponent uses 2*pi*sigma^4
                const double Gy = -y * exp(-r2 / (2.0 * PI * s * s * s * s));
                acc += cos(th) * Gx + sin(th) * Gy;
            } else {
                const double s2 = s * s;
                const double et = exp(-r2 / (2.0 * s2));
                const double Gxx = -1.0 / (2.0 * PI * s2 * s2) * (1.0 - x * x / (s2 * s2)) * et;
                const double Gyy = -1.0 / (2.0 * PI * s2 * s2) * (1.0 - y * y / (s2 * s2)) * et;
                const double Gxy = x * y / (2.0 * PI * s2 * s2 * s2) * et;
                const double cc = cos(th), ss = sin(th);
                acc += cc * cc * Gxx + ss * ss * (Gyy - Gxy) - 2.0 * ss * cc * (Gyy + Gxy);
            }
        }
        W12[c][ai][bi] = (float)acc;
    }

    // ---- sum of dilated 3x3 Gaussians (rates 1,2,3), embedded in 7x7 ----
    for (int e = tid; e < 49; e += 256) {
        const int u = e / 7, v = e % 7;
        double g[3][3], gs = 0.0;
        for (int i = 0; i < 3; ++i)
            for (int j = 0; j < 3; ++j) {
                const double xx = i - 1, yy = j - 1;
                g[i][j] = exp(-(xx * xx + yy * yy) / 2.0) / (2.0 * PI);
                gs += g[i][j];
            }
        double acc = 0.0;
        for (int d = 1; d <= 3; ++d) {
            const int p = 3 - d;
            const int iu = u - p, iv = v - p;
            if (iu >= 0 && iv >= 0 && iu % d == 0 && iv % d == 0 &&
                iu / d < 3 && iv / d < 3)
                acc += g[iu / d][iv / d] / gs;
        }
        G7[u][v] = (float)acc;
    }
    __syncthreads();

    // ---- composite 13x13 = full conv(W, G); pack f16 A [16][224] ----
    for (int e = tid; e < 16 * KPAD; e += 256) {
        const int m = e / KPAD, k = e % KPAD;
        const int dy = k >> 4, dx = k & 15;
        float val = 0.f;
        if (m < NCH && dy < 13 && dx < 13) {
            float acc = 0.f;
            for (int i = 0; i < 7; ++i)
                for (int j = 0; j < 7; ++j) {
                    const int gi = dy - i, gj = dx - j;
                    if (gi >= 0 && gi < 7 && gj >= 0 && gj < 7)
                        acc += W12[m][i][j] * G7[gi][gj];
                }
            val = acc;
        }
        Apack[e] = (_Float16)val;
    }
}

// ---------------------------------------------------------------------------
// K2: RGB -> grayscale f16, float4-vectorized
// ---------------------------------------------------------------------------
__global__ __launch_bounds__(256)
void gray_kernel(const float* __restrict__ img, _Float16* __restrict__ gray)
{
    const int p = blockIdx.x * blockDim.x + threadIdx.x;   // float4 index
    const int b = p >> 16;                                 // 65536 float4 / plane
    const int o = p & 65535;
    const size_t plane = (size_t)IMG * IMG;
    const float4 r  = ((const float4*)(img + (size_t)(b * 3 + 0) * plane))[o];
    const float4 g  = ((const float4*)(img + (size_t)(b * 3 + 1) * plane))[o];
    const float4 bl = ((const float4*)(img + (size_t)(b * 3 + 2) * plane))[o];
    const float s = 1.0f / 255.0f;
    v4h h;
    h[0] = (_Float16)((0.299f * r.x + 0.587f * g.x + 0.114f * bl.x) * s);
    h[1] = (_Float16)((0.299f * r.y + 0.587f * g.y + 0.114f * bl.y) * s);
    h[2] = (_Float16)((0.299f * r.z + 0.587f * g.z + 0.114f * bl.z) * s);
    h[3] = (_Float16)((0.299f * r.w + 0.587f * g.w + 0.114f * bl.w) * s);
    ((v4h*)gray)[p] = h;
}

// ---------------------------------------------------------------------------
// K3: implicit-GEMM conv via v_wmma_f32_16x16x32_f16.
//     M=16 (12 ch + 4 zero rows), N=16 pixels, K=224 (k = dy*16+dx, padded).
//
//     A layout (16-bit A 16x32, contiguous k order): lane l holds row M=l%16;
//       VGPR v holds K pair hi*16 + v*2 within the chunk (same order as B).
//     B layout (32x16): lane half hi covers chunk K hi*16..hi*16+15, i.e.
//       tap row dy = 2q+hi -> 16 contiguous f16 at a per-chunk-constant base.
//     4 independent accumulator chains (one per output row) hide WMMA latency.
// ---------------------------------------------------------------------------
__global__ __launch_bounds__(128)
void conv_wmma(const _Float16* __restrict__ gray, const _Float16* __restrict__ Apack,
               float* __restrict__ out, float* __restrict__ stats)
{
    __shared__ _Float16 tile[RH * RW];
    __shared__ float bsum[NCH * 2];

    const int img = blockIdx.z;
    const int x0 = blockIdx.x * TILE, y0 = blockIdx.y * TILE;
    const int tid = threadIdx.x;
    const int lane = tid & 31, wave = tid >> 5;
    const int hi = (lane >> 4) & 1;      // half-wave: selects K/M sub-blocks
    const int l16 = lane & 15;           // A row (channel) == B/D column (pixel)

    if (tid < NCH * 2) bsum[tid] = 0.f;

    // ---- cooperative halo load (zero outside image; row 28 forced zero) ----
    const _Float16* gimg = gray + (size_t)img * IMG * IMG;
    for (int i = tid; i < RH * RW; i += 128) {
        const int ry = i / RW, rx = i % RW;
        const int gy = y0 - HALO + ry, gx = x0 - HALO + rx;
        _Float16 v = (_Float16)0.f;
        if (ry < 28 && gy >= 0 && gy < IMG && gx >= 0 && gx < IMG)
            v = gimg[gy * IMG + gx];
        tile[i] = v;
    }
    __syncthreads();

    // ---- A fragments: 7 K-chunks of the 16x224 weight matrix ----
    union AU { v16h h; unsigned u[8]; };
    AU a[NQ];
    const unsigned* Au = (const unsigned*)Apack;       // [16][112] dwords
    #pragma unroll
    for (int q = 0; q < NQ; ++q) {
        #pragma unroll
        for (int v = 0; v < 8; ++v) {
            // dword index of K pair: K = q*32 + hi*16 + v*2
            const int dw = q * 16 + hi * 8 + v;
            a[q].u[v] = Au[l16 * (KPAD / 2) + dw];
        }
    }

    const int row0 = wave * 4;                         // this wave: rows row0..row0+3
    v8f c[4] = {};

    #pragma unroll
    for (int q = 0; q < NQ; ++q) {
        const int dy = 2 * q + hi;                     // B tap row for this half-wave
        #pragma unroll
        for (int rr = 0; rr < 4; ++rr) {
            // B fragment: lane l holds column N=l%16, K j -> tap (dy, dx=j):
            // 16 contiguous f16 starting at tile[(row+dy)*RW + l16].
            union BU { v16h h; _Float16 e[16]; } b;
            const _Float16* rowp = &tile[(row0 + rr + dy) * RW + l16];
            #pragma unroll
            for (int j = 0; j < 16; ++j) b.e[j] = rowp[j];
            c[rr] = __builtin_amdgcn_wmma_f32_16x16x32_f16(
                        false, a[q].h, false, b.h, (short)0, c[rr], false, false);
        }
    }

    // ---- store + per-(b,c) sum / sumsq ----
    float accS[8] = {}, accQ[8] = {};
    #pragma unroll
    for (int rr = 0; rr < 4; ++rr) {
        const int gy = y0 + row0 + rr, gx = x0 + l16;
        #pragma unroll
        for (int r = 0; r < 8; ++r) {
            const int m = r + hi * 8;     // D layout: lanes 0-15 M=r, 16-31 M=8+r
            const float v = c[rr][r];
            if (m < NCH)
                out[(((size_t)img * NCH + m) * IMG + gy) * IMG + gx] = v;
            accS[r] += v;                 // channels >= 12 are exactly 0
            accQ[r] += v * v;
        }
    }

    #pragma unroll
    for (int r = 0; r < 8; ++r) {
        float s = accS[r], q = accQ[r];
        for (int off = 1; off < 16; off <<= 1) {
            s += __shfl_xor(s, off, 32);
            q += __shfl_xor(q, off, 32);
        }
        if (l16 == 0) {
            const int m = r + hi * 8;
            if (m < NCH) {
                atomicAdd(&bsum[m * 2 + 0], s);
                atomicAdd(&bsum[m * 2 + 1], q);
            }
        }
    }
    __syncthreads();
    if (tid < NCH * 2) atomicAdd(&stats[img * NCH * 2 + tid], bsum[tid]);
}

// ---------------------------------------------------------------------------
// K4: in-place per-(b,c) normalization with unbiased std, float4-vectorized
// ---------------------------------------------------------------------------
__global__ __launch_bounds__(256)
void norm_kernel(float* __restrict__ out, const float* __restrict__ stats)
{
    const int p = blockIdx.x * blockDim.x + threadIdx.x;   // float4 index
    const int ci = p >> 16;                                // 65536 float4 / plane
    const float N = (float)(IMG * IMG);
    const float s = stats[ci * 2 + 0];
    const float q = stats[ci * 2 + 1];
    const float mean = s / N;
    float var = (q - s * s / N) / (N - 1.f);
    var = var > 0.f ? var : 0.f;
    const float inv = 1.f / (sqrtf(var) + 1e-6f);
    float4 v = ((const float4*)out)[p];
    v.x = (v.x - mean) * inv;
    v.y = (v.y - mean) * inv;
    v.z = (v.z - mean) * inv;
    v.w = (v.w - mean) * inv;
    ((float4*)out)[p] = v;
}

// ---------------------------------------------------------------------------
extern "C" void kernel_launch(void* const* d_in, const int* in_sizes, int n_in,
                              void* d_out, int out_size, void* d_ws, size_t ws_size,
                              hipStream_t stream)
{
    const float* image = (const float*)d_in[0];
    float* out = (float*)d_out;
    char* ws = (char*)d_ws;

    // workspace layout
    _Float16* Apack = (_Float16*)ws;                 //  7 KB  : f16 [16][224]
    float*    stats = (float*)(ws + 8192);           //  1.5 KB: [16][12][2]
    _Float16* gray  = (_Float16*)(ws + 16384);       //  8 MB  : f16 [16][512][512]

    init_kernel<<<1, 256, 0, stream>>>(Apack, stats);

    gray_kernel<<<(NB * IMG * IMG / 4) / 256, 256, 0, stream>>>(image, gray);

    dim3 grid(IMG / TILE, IMG / TILE, NB);           // 32 x 32 x 16
    conv_wmma<<<grid, 128, 0, stream>>>(gray, Apack, out, stats);

    norm_kernel<<<(NB * NCH * IMG * IMG / 4) / 256, 256, 0, stream>>>(out, stats);
}
